// Attention_61598420959496
// MI455X (gfx1250) — matrix operand order repaired and checked by reading, not compiled, not run on hardware
//
#include <hip/hip_runtime.h>

typedef __attribute__((ext_vector_type(16))) _Float16 v16h;
typedef __attribute__((ext_vector_type(8)))  _Float16 v8h;
typedef __attribute__((ext_vector_type(8)))  float    v8f;
typedef __attribute__((ext_vector_type(4)))  unsigned int v4u;
typedef __attribute__((ext_vector_type(8)))  int v8i;
typedef __attribute__((ext_vector_type(4)))  int v4i;

#define D_MODEL 1024
#define N_HEADS 16
#define DK      64
#define SEQ     1024
#define BATCH   4
#define MTOT    (BATCH * SEQ)

__device__ __forceinline__ v16h make_frag(v8h a, v8h b) {
  v16h r;
#pragma unroll
  for (int i = 0; i < 8; ++i) { r[i] = a[i]; r[i + 8] = b[i]; }
  return r;
}

// A fragment (16x32, MxK) from row-major [16][ld] at p (ld in halves).
__device__ __forceinline__ v16h load_a_frag(const _Float16* p, int ld, int lane) {
  const int m = lane & 15, h = lane >> 4;
  const _Float16* row = p + m * ld + 16 * h;
  return make_frag(*(const v8h*)row, *(const v8h*)(row + 8));
}

// B fragment (32x16, KxN) from K-contiguous storage: elem (k,n) at pT[n*ld + k].
__device__ __forceinline__ v16h load_bT_frag(const _Float16* pT, int ld, int lane) {
  const int n = lane & 15, h = lane >> 4;
  const _Float16* base = pT + n * ld + 16 * h;
  return make_frag(*(const v8h*)base, *(const v8h*)(base + 8));
}

__device__ __forceinline__ v8f wmma32(v16h a, v16h b, v8f c) {
  return __builtin_amdgcn_wmma_f32_16x16x32_f16(false, a, false, b, (short)0, c, false, false);
}

// ---- Tensor Data Mover: one descriptor-driven DMA per tile (TENSORcnt-tracked) ----
// 2-D D# per CDNA5 ISA:
//   group0: count=1 | lds_addr | global_addr | type=2
//   group1: data_size=2B, tensor_dim0/1, tile_dim0/1, tensor_dim0_stride
//   groups 2/3: zero (2-D tensor)
__device__ __forceinline__ void tdm_load_tile_f16(unsigned lds_addr, const _Float16* gsrc,
                                                  unsigned tile_rows, unsigned tile_cols,
                                                  unsigned tensor_rows, unsigned tensor_cols,
                                                  unsigned row_stride_elems) {
  const unsigned long long ga = (unsigned long long)(uintptr_t)gsrc;
  v4u g0;
  g0[0] = 1u;                                                   // one valid descriptor
  g0[1] = lds_addr;                                             // LDS byte address
  g0[2] = (unsigned)ga;                                         // global_addr[31:0]
  g0[3] = ((unsigned)(ga >> 32) & 0x01FFFFFFu) | (2u << 30);    // global_addr[56:32] | type=2
  v8i g1;
  g1[0] = (int)(1u << 16);                                      // workgroup_mask=0, data_size=1 (2B)
  g1[1] = (int)((tensor_cols & 0xFFFFu) << 16);                 // tensor_dim0 lo16 (bits 63:48)
  g1[2] = (int)(((tensor_cols >> 16) & 0xFFFFu) | ((tensor_rows & 0xFFFFu) << 16));
  g1[3] = (int)(((tensor_rows >> 16) & 0xFFFFu) | ((tile_cols & 0xFFFFu) << 16));
  g1[4] = (int)(tile_rows & 0xFFFFu);                           // tile_dim1 (tile_dim2 = 0)
  g1[5] = (int)row_stride_elems;                                // tensor_dim0_stride lo32
  g1[6] = 0;
  g1[7] = 0;
  const v4i gz = {0, 0, 0, 0};
#if __clang_major__ >= 23
  const v8i gz8 = {0, 0, 0, 0, 0, 0, 0, 0};
  __builtin_amdgcn_tensor_load_to_lds(g0, g1, gz, gz, gz8, 0);
#else
  __builtin_amdgcn_tensor_load_to_lds(g0, g1, gz, gz, 0);
#endif
}

__global__ void cvt_f32_f16(const float* __restrict__ in, _Float16* __restrict__ out, int n) {
  int i = blockIdx.x * blockDim.x + threadIdx.x;
  if (i < n) out[i] = (_Float16)in[i];
}

// Weight [K=1024][N=1024] f32 -> transposed f16 [N][K] (one-off).
__global__ void cvt_w_t(const float* __restrict__ in, _Float16* __restrict__ out) {
  int i = blockIdx.x * blockDim.x + threadIdx.x;
  int k = i >> 10, n = i & 1023;
  out[n * D_MODEL + k] = (_Float16)in[i];
}

// C = A[M x 1024] @ W + bias (W pre-transposed Wt[n][k]).
// 128x128 tile per 256-thread block (8 waves). Tiles staged by TDM, double-buffered.
// MODE 0: f16 out [B][H][S][64] (Q,K);  MODE 1: f16 out [B][H][64][S] (V^T);
// MODE 2: f32 out [M][1024] (final projection).
template <int MODE>
__global__ __launch_bounds__(256) void gemm16(const _Float16* __restrict__ A,
                                              const _Float16* __restrict__ Wt,
                                              const float* __restrict__ bias,
                                              void* __restrict__ outp) {
  __shared__ __align__(16) _Float16 lA[2][128 * 32];   // [m][k], ld=32
  __shared__ __align__(16) _Float16 lBt[2][128 * 32];  // [n][k], ld=32
  const int m0 = blockIdx.x * 128;
  const int n0 = blockIdx.y * 128;
  const int t = threadIdx.x;
  const int wave = t >> 5, lane = t & 31;
  const bool issuer = (wave == 0);   // TDM issues once per wave -> single-wave guard

  const _Float16* ga = A  + (size_t)m0 * D_MODEL;
  const _Float16* gb = Wt + (size_t)n0 * D_MODEL;

  v8f acc[8] = {};

  if (issuer) {
    tdm_load_tile_f16((unsigned)(uintptr_t)&lA[0][0],  ga, 128, 32, MTOT,    D_MODEL, D_MODEL);
    tdm_load_tile_f16((unsigned)(uintptr_t)&lBt[0][0], gb, 128, 32, D_MODEL, D_MODEL, D_MODEL);
  }

  for (int it = 0; it < D_MODEL / 32; ++it) {
    const int cur = it & 1;
    if (issuer) {
      if (it + 1 < D_MODEL / 32) {   // prefetch next k-tile into the other buffer
        tdm_load_tile_f16((unsigned)(uintptr_t)&lA[cur ^ 1][0],  ga + (it + 1) * 32, 128, 32,
                          MTOT, D_MODEL, D_MODEL);
        tdm_load_tile_f16((unsigned)(uintptr_t)&lBt[cur ^ 1][0], gb + (it + 1) * 32, 128, 32,
                          D_MODEL, D_MODEL, D_MODEL);
        __builtin_amdgcn_s_wait_tensorcnt((short)2);  // in-order: current pair has landed
      } else {
        __builtin_amdgcn_s_wait_tensorcnt((short)0);
      }
    }
    __syncthreads();
    v16h af = load_a_frag(&lA[cur][(16 * wave) * 32], 32, lane);
#pragma unroll
    for (int j = 0; j < 8; ++j) {
      v16h bf = load_bT_frag(&lBt[cur][(16 * j) * 32], 32, lane);
      acc[j] = wmma32(af, bf, acc[j]);
    }
    __syncthreads();
  }

#pragma unroll
  for (int j = 0; j < 8; ++j) {
    const int n = n0 + 16 * j + (lane & 15);
    const float bn = bias[n];
#pragma unroll
    for (int r = 0; r < 8; ++r) {
      const int srow = m0 + 16 * wave + r + 8 * (lane >> 4);
      const float v = acc[j][r] + bn;
      if (MODE == 2) {
        ((float*)outp)[srow * D_MODEL + n] = v;
      } else {
        const int b = srow >> 10, sl = srow & 1023;
        const int h = n >> 6, d = n & 63;
        if (MODE == 0)
          ((_Float16*)outp)[(((b * N_HEADS + h) * SEQ + sl) << 6) + d] = (_Float16)v;
        else
          ((_Float16*)outp)[(((b * N_HEADS + h) * DK + d) << 10) + sl] = (_Float16)v;
      }
    }
  }
}

// Flash attention, transposed-score formulation: one wave per (b, h, 16-query tile).
// The 4 waves of a block share (b, h) and take consecutive q-tiles -> identical
// K/V cachelines per iteration (WGP$/L2 reuse).
// S^T = K·Q^T (m = key, n = query): softmax rows live per lane-pair.  O^T = V^T·P^T.
__global__ __launch_bounds__(128) void attn(const _Float16* __restrict__ Q,
                                            const _Float16* __restrict__ K,
                                            const _Float16* __restrict__ Vt,
                                            _Float16* __restrict__ Ao) {
  const int t = threadIdx.x, wave = t >> 5, lane = t & 31;
  const int bh = blockIdx.x >> 4;               // b*16 + h (64 values)
  const int qt = (blockIdx.x & 15) * 4 + wave;  // 64 q-tiles per head
  const int hd = bh & 15;
  const int b  = bh >> 4;
  const _Float16* Qb = Q + (size_t)bh * SEQ * DK;
  const _Float16* Kb = K + (size_t)bh * SEQ * DK;
  const _Float16* Vb = Vt + (size_t)bh * DK * SEQ;
  const int q0 = qt * 16;
  const int nloc = lane & 15, hh = lane >> 4;

  // Q^T B-fragments (loop-invariant): element (k=d, n=q) contiguous along d.
  const v16h bq0 = load_bT_frag(Qb + q0 * DK,      DK, lane);
  const v16h bq1 = load_bT_frag(Qb + q0 * DK + 32, DK, lane);

  float mcur = -3.0e38f, lcur = 0.0f;
  v8f ot[4] = {};                    // O^T accum: (m = d tile jd, n = q)
  const float scale = 0.125f;        // 1/sqrt(64)

  for (int kk = 0; kk < SEQ; kk += 32) {
    v8f s[2] = {};
#pragma unroll
    for (int j = 0; j < 2; ++j) {    // S^T tiles: key rows kk+16j..+15
      v16h ka0 = load_a_frag(Kb + (kk + 16 * j) * DK,      DK, lane);
      v16h ka1 = load_a_frag(Kb + (kk + 16 * j) * DK + 32, DK, lane);
      s[j] = wmma32(ka0, bq0, s[j]);
      s[j] = wmma32(ka1, bq1, s[j]);
    }
    // Per-query softmax: lane pair (L, L+16) covers all 32 keys of this tile.
    float v0[8], v1[8];
    float lm = -3.0e38f;
#pragma unroll
    for (int r = 0; r < 8; ++r) {
      v0[r] = s[0][r] * scale;
      v1[r] = s[1][r] * scale;
      lm = fmaxf(lm, fmaxf(v0[r], v1[r]));
    }
    lm = fmaxf(lm, __shfl_xor(lm, 16, 32));
    const float mn = fmaxf(mcur, lm);
    const float corr = __expf(mcur - mn);
    float p0[8], p1[8], ls = 0.0f;
#pragma unroll
    for (int r = 0; r < 8; ++r) {
      p0[r] = __expf(v0[r] - mn);
      p1[r] = __expf(v1[r] - mn);
      ls += p0[r] + p1[r];
    }
    ls += __shfl_xor(ls, 16, 32);
    lcur = lcur * corr + ls;
    mcur = mn;
#pragma unroll
    for (int jd = 0; jd < 4; ++jd)
#pragma unroll
      for (int r = 0; r < 8; ++r) ot[jd][r] *= corr;

    // Build P^T B-fragment (k = key-in-tile, n = q): half-swap between lane pairs.
    v16h pf;
#pragma unroll
    for (int r = 0; r < 8; ++r) {
      const float sw0 = __shfl_xor(p0[r], 16, 32);
      const float sw1 = __shfl_xor(p1[r], 16, 32);
      pf[r]     = (_Float16)(hh ? sw1 : p0[r]);   // e=0..7 : k = 16h + r
      pf[8 + r] = (_Float16)(hh ? p1[r] : sw0);   // e=8..15: k = 16h + 8 + r
    }
#pragma unroll
    for (int jd = 0; jd < 4; ++jd) {   // O^T += V^T(tile jd) x P^T
      v16h va = load_a_frag(Vb + (16 * jd) * SEQ + kk, SEQ, lane);
      ot[jd] = wmma32(va, pf, ot[jd]);
    }
  }

  // Normalize and store: for fixed lane (fixed q), vgpr index runs contiguously in d.
  const float inv = 1.0f / lcur;
  const int srow = b * SEQ + q0 + nloc;
#pragma unroll
  for (int jd = 0; jd < 4; ++jd) {
    v8h ov;
#pragma unroll
    for (int r = 0; r < 8; ++r) ov[r] = (_Float16)(ot[jd][r] * inv);
    *(v8h*)(Ao + (size_t)srow * D_MODEL + hd * DK + 16 * jd + 8 * hh) = ov;
  }
}

extern "C" void kernel_launch(void* const* d_in, const int* in_sizes, int n_in,
                              void* d_out, int out_size, void* d_ws, size_t ws_size,
                              hipStream_t stream) {
  const float* queries = (const float*)d_in[0];
  const float* keys    = (const float*)d_in[1];
  const float* values  = (const float*)d_in[2];
  const float* Wq = (const float*)d_in[3];
  const float* bq = (const float*)d_in[4];
  const float* Wk = (const float*)d_in[5];
  const float* bk = (const float*)d_in[6];
  const float* Wv = (const float*)d_in[7];
  const float* bv = (const float*)d_in[8];
  const float* Wo = (const float*)d_in[9];
  const float* bo = (const float*)d_in[10];

  char* ws = (char*)d_ws;
  size_t off = 0;
  auto alloc = [&](size_t bytes) {
    char* p = ws + off;
    off += (bytes + 255) & ~(size_t)255;
    return p;
  };
  const size_t actB = (size_t)MTOT * D_MODEL * sizeof(_Float16);    // 8 MiB
  const size_t wB   = (size_t)D_MODEL * D_MODEL * sizeof(_Float16); // 2 MiB
  _Float16* xq = (_Float16*)alloc(actB);
  _Float16* xk = (_Float16*)alloc(actB);
  _Float16* xv = (_Float16*)alloc(actB);
  _Float16* wqt = (_Float16*)alloc(wB);   // transposed weights [n][k]
  _Float16* wkt = (_Float16*)alloc(wB);
  _Float16* wvt = (_Float16*)alloc(wB);
  _Float16* wot = (_Float16*)alloc(wB);
  _Float16* qw = (_Float16*)alloc(actB);  // [B][H][S][64]
  _Float16* kw = (_Float16*)alloc(actB);  // [B][H][S][64]
  _Float16* vt = (_Float16*)alloc(actB);  // [B][H][64][S]
  _Float16* ao = (_Float16*)alloc(actB);  // [B*S][H*64]

  const int nAct = MTOT * D_MODEL;
  const int nW   = D_MODEL * D_MODEL;
  cvt_f32_f16<<<nAct / 256, 256, 0, stream>>>(queries, xq, nAct);
  cvt_f32_f16<<<nAct / 256, 256, 0, stream>>>(keys,    xk, nAct);
  cvt_f32_f16<<<nAct / 256, 256, 0, stream>>>(values,  xv, nAct);
  cvt_w_t<<<nW / 256, 256, 0, stream>>>(Wq, wqt);
  cvt_w_t<<<nW / 256, 256, 0, stream>>>(Wk, wkt);
  cvt_w_t<<<nW / 256, 256, 0, stream>>>(Wv, wvt);
  cvt_w_t<<<nW / 256, 256, 0, stream>>>(Wo, wot);

  dim3 ggrid(MTOT / 128, D_MODEL / 128);
  gemm16<0><<<ggrid, 256, 0, stream>>>(xq, wqt, bq, (void*)qw);
  gemm16<0><<<ggrid, 256, 0, stream>>>(xk, wkt, bk, (void*)kw);
  gemm16<1><<<ggrid, 256, 0, stream>>>(xv, wvt, bv, (void*)vt);

  // 64 (b,h) pairs x 16 groups of 4 q-tile waves
  attn<<<1024, 128, 0, stream>>>(qw, kw, vt, ao);

  gemm16<2><<<ggrid, 256, 0, stream>>>(ao, wot, bo, d_out);
}
